// DotAttentionV4_72378788872877
// MI455X (gfx1250) — matrix operand order repaired and checked by reading, not compiled, not run on hardware
//
#include <hip/hip_runtime.h>
#include <hip/hip_bf16.h>

// Problem constants (match reference)
#define NB 256   // batch
#define NN 128   // sequence
#define ND 64    // head dim

typedef __attribute__((ext_vector_type(16))) __bf16 v16bf;
typedef __attribute__((ext_vector_type(8)))  float  v8f;

// Exact pointee type expected by the async-to-LDS builtin (from hipcc diag):
// 'int __attribute__((vector_size(16)))*'
typedef int v4i __attribute__((vector_size(16)));
typedef __attribute__((address_space(1))) v4i* gptr4;
typedef __attribute__((address_space(3))) v4i* lptr4;

// l0 shift: -BETA*log(-GAMMA/ZETA) = -(2/3)*ln(0.1/1.1) = +1.5985968485
#define L0_SHIFT 1.5985968485f

#if __has_builtin(__builtin_amdgcn_global_load_async_to_lds_b128)
#define HAVE_ASYNC_LDS 1
#else
#define HAVE_ASYNC_LDS 0
#endif

// Native bf16 convert (gfx1250 has bf16 VALU; single v_cvt instead of bfe/add3 RNE)
__device__ __forceinline__ unsigned short f2bf(float f) {
  return __builtin_bit_cast(unsigned short, (__bf16)f);
}
__device__ __forceinline__ float fsigmoid(float x) {
  return 1.0f / (1.0f + __expf(-x));
}
__device__ __forceinline__ v8f vzero() {
  v8f z;
#pragma unroll
  for (int i = 0; i < 8; ++i) z[i] = 0.0f;
  return z;
}

__device__ __forceinline__ void wait_async_le4() {
#if __has_builtin(__builtin_amdgcn_s_wait_asynccnt)
  __builtin_amdgcn_s_wait_asynccnt(4);
#else
  asm volatile("s_wait_asynccnt 0x4" ::: "memory");
#endif
}
__device__ __forceinline__ void wait_async_0() {
#if __has_builtin(__builtin_amdgcn_s_wait_asynccnt)
  __builtin_amdgcn_s_wait_asynccnt(0);
#else
  asm volatile("s_wait_asynccnt 0x0" ::: "memory");
#endif
}

// Issue a 16 KB tile copy global->LDS, 16 B per lane x 4 issues (ASYNCcnt += 4),
// or a synchronous fallback. Each thread later consumes exactly its own bytes.
__device__ __forceinline__ void issue_tile_copy(const float* __restrict__ zt,
                                                unsigned char* braw, int tid) {
#pragma unroll
  for (int c = 0; c < 4; ++c) {
    const float* gp = zt + c * 1024 + tid * 4;
    unsigned char* lp = braw + c * 4096 + tid * 16;
#if HAVE_ASYNC_LDS
    __builtin_amdgcn_global_load_async_to_lds_b128(
        (gptr4)(v4i*)const_cast<float*>(gp), (lptr4)(v4i*)lp, 0, 0);
#else
    *(float4*)lp = *(const float4*)gp;
#endif
  }
}

union AFrag { unsigned int u[8]; v16bf v; };

// A fragment (16x32 bf16, M=row striped in lane, §7.12.2): lane's 16 K-values are
// [k0 + hi*8, +8) and [k0 + 16 + hi*8, +8) of its row. p = &row[k0]; 16B aligned.
__device__ __forceinline__ v16bf load_a(const unsigned short* p, int hi) {
  AFrag f;
  uint4 lo = *(const uint4*)(p + hi * 8);
  uint4 hh = *(const uint4*)(p + 16 + hi * 8);
  f.u[0] = lo.x; f.u[1] = lo.y; f.u[2] = lo.z; f.u[3] = lo.w;
  f.u[4] = hh.x; f.u[5] = hh.y; f.u[6] = hh.z; f.u[7] = hh.w;
  return f.v;
}
// B fragment (32x16 bf16, N=col striped across lanes): lane's 16 K-values are
// [k0 + hi*8, +8) and [k0 + 16 + hi*8, +8) of its column (stored K-contiguous in LDS). 16B aligned.
__device__ __forceinline__ v16bf load_b(const unsigned short* p, int hi) {
  AFrag f;
  uint4 lo = *(const uint4*)(p + hi * 8);
  uint4 hh = *(const uint4*)(p + 16 + hi * 8);
  f.u[0] = lo.x; f.u[1] = lo.y; f.u[2] = lo.z; f.u[3] = lo.w;
  f.u[4] = hh.x; f.u[5] = hh.y; f.u[6] = hh.z; f.u[7] = hh.w;
  return f.v;
}

__global__ void init_l0_kernel(float* l0) {
  if (threadIdx.x == 0) *l0 = 0.0f;
}

// ---------------------------------------------------------------------------
// Kernel 1: hard-concrete gate + l0 + bilinear scores.
// Grid: (N/4)*(N/4) = 1024 blocks, 256 threads (8 waves).
// Z[i,j] tiles stream via async-to-LDS double buffer (ASYNCcnt pipelined);
// gate+l0 fused into the single pass; scores via WMMA:
//   U = Z*K_tile^T ; S = Q_tile*U ; take diag.
// ---------------------------------------------------------------------------
__global__ __launch_bounds__(256)
void score_gate_kernel(const float* __restrict__ q,
                       const float* __restrict__ k,
                       const float* __restrict__ mat,
                       float* __restrict__ scores,   // [i][j][b]
                       float* __restrict__ l0_out) {
  extern __shared__ unsigned char smem[];
  unsigned char*  z_raw = smem;                       // 2 x 16 KB raw Z dbl-buf
  unsigned short* q_s = (unsigned short*)(smem + 32768);  // 4 panels [b][d] :128 KB
  unsigned short* k_s = q_s + 4 * NB * ND;            // 4 panels [b][d]     :128 KB
  unsigned short* z_s = k_s + 4 * NB * ND;            // 64x64 bf16 gated    :  8 KB
  unsigned short* u_s = z_s + ND * ND;                // per-wave 16x64 bf16 : 16 KB
  float*          red = (float*)(u_s + 8 * 16 * ND);  // 8 floats (l0 reduce)

  const int tid  = threadIdx.x;
  const int wave = tid >> 5;
  const int lane = tid & 31;
  const int n16  = lane & 15;
  const int hi   = lane >> 4;

  const int i0 = (blockIdx.x >> 5) * 4;
  const int j0 = (blockIdx.x & 31) * 4;
#define ZTILE(n) (mat + ((size_t)((i0 + ((n) >> 2)) * NN + j0 + ((n) & 3))) * (ND * ND))

  // Kick off the async pipeline for pair 0 before staging q/k.
  issue_tile_copy(ZTILE(0), z_raw, tid);

  // Stage q/k panels f32 -> bf16 (coalesced; L2-resident source).
  for (int p = 0; p < 4; ++p) {
    for (int idx = tid; idx < NB * ND; idx += 256) {
      const int b = idx >> 6, d = idx & 63;
      q_s[p * NB * ND + idx] = f2bf(q[((size_t)b * NN + (i0 + p)) * ND + d]);
      k_s[p * NB * ND + idx] = f2bf(k[((size_t)b * NN + (j0 + p)) * ND + d]);
    }
  }

  float l0_acc = 0.0f;

  for (int n = 0; n < 16; ++n) {
    const int pi = n >> 2, pj = n & 3;
    const int i = i0 + pi, j = j0 + pj;

    // Issue next tile into the other buffer, then wait for the current one.
    if (n + 1 < 16) {
      issue_tile_copy(ZTILE(n + 1), z_raw + ((n + 1) & 1) * 16384, tid);
      // Warm L2 two tiles ahead (global_prefetch_b8, speculative).
      if (n + 2 < 16) __builtin_prefetch(ZTILE(n + 2) + tid * 16, 0, 1);
      wait_async_le4();
    } else {
      wait_async_0();
    }

    const unsigned char* braw = z_raw + (n & 1) * 16384;

    __syncthreads();  // previous pair's WMMA readers done with z_s
    // Gate + l0 fused into the one-and-only pass over this 16 KB tile.
#pragma unroll
    for (int c = 0; c < 4; ++c) {
      const int idx = c * 1024 + tid * 4;
      const float4 xv = *(const float4*)(braw + idx * 4);
      const float xs[4] = {xv.x, xv.y, xv.z, xv.w};
#pragma unroll
      for (int e = 0; e < 4; ++e) {
        const float sg = fsigmoid(xs[e]);
        const float zz = fminf(fmaxf(sg * 1.2f - 0.1f, 0.0f), 1.0f);
        z_s[idx + e] = f2bf(zz);
        l0_acc += fsigmoid(xs[e] + L0_SHIFT);
      }
    }
    __syncthreads();

    // Each wave: 2 batch-tiles of 16.
#pragma unroll
    for (int t = 0; t < 2; ++t) {
      const int b0 = (wave + t * 8) * 16;

      // U(64d x 16b) = Z(64x64) @ K_tile^T(64e x 16b)
      v8f uacc[4];
#pragma unroll
      for (int dt = 0; dt < 4; ++dt) uacc[dt] = vzero();
#pragma unroll
      for (int es = 0; es < 2; ++es) {
        const int e0 = es * 32;
        const v16bf bfr =
            load_b(&k_s[((size_t)pj * NB + b0 + n16) * ND + e0], hi);
#pragma unroll
        for (int dt = 0; dt < 4; ++dt) {
          const v16bf afr = load_a(&z_s[(dt * 16 + n16) * ND + e0], hi);
          uacc[dt] = __builtin_amdgcn_wmma_f32_16x16x32_bf16(
              false, afr, false, bfr, (short)0, uacc[dt], false, false);
        }
      }
      // Spill U transposed to LDS as bf16: u_s[wave][b_col][d]
      unsigned short* uw = u_s + wave * 16 * ND;
#pragma unroll
      for (int dt = 0; dt < 4; ++dt)
#pragma unroll
        for (int r = 0; r < 8; ++r)
          uw[n16 * ND + dt * 16 + r + hi * 8] = f2bf(uacc[dt][r]);

      // S16(16b x 16b) = Q_tile(16b x 64d) @ U(64d x 16b); we need its diag.
      v8f sacc = vzero();
#pragma unroll
      for (int dsr = 0; dsr < 2; ++dsr) {
        const int d0 = dsr * 32;
        const v16bf afr =
            load_a(&q_s[((size_t)pi * NB + b0 + n16) * ND + d0], hi);
        const v16bf bfr = load_b(&uw[n16 * ND + d0], hi);
        sacc = __builtin_amdgcn_wmma_f32_16x16x32_bf16(
            false, afr, false, bfr, (short)0, sacc, false, false);
      }
      // Diag via per-wave LDS bounce; alias the (consumed) U region.
      // In-wave DS ordering makes store-after-operand-load safe.
      float* sw = (float*)uw;  // 2 KB >= 16x16 f32
#pragma unroll
      for (int r = 0; r < 8; ++r) sw[(r + hi * 8) * 16 + n16] = sacc[r];
      if (lane < 16) {
        const float sv = sw[lane * 16 + lane] * 0.125f;  // 1/sqrt(64)
        scores[((size_t)(i * NN + j)) * NB + b0 + lane] = sv;  // 64B coalesced
      }
    }
  }
#undef ZTILE

  // Block-reduce l0 and accumulate globally.
#pragma unroll
  for (int o = 16; o >= 1; o >>= 1) l0_acc += __shfl_xor(l0_acc, o, 32);
  if (lane == 0) red[wave] = l0_acc;
  __syncthreads();
  if (tid == 0) {
    float s = 0.0f;
#pragma unroll
    for (int w = 0; w < 8; ++w) s += red[w];
    atomicAdd(l0_out, s);
  }
}

// ---------------------------------------------------------------------------
// Kernel 2: per-batch softmax + attn @ V via WMMA.
// Grid: 256 blocks (one per b), 256 threads (8 waves).
// ---------------------------------------------------------------------------
__global__ __launch_bounds__(256)
void softmax_av_kernel(const float* __restrict__ scores,  // [i][j][b]
                       const float* __restrict__ v,
                       float* __restrict__ out) {
  __shared__ unsigned short a_s[NN * NN];  // attn bf16 [i][j]         32 KB
  __shared__ unsigned short v_s[ND * NN];  // V bf16 transposed [d][j] 16 KB

  const int tid  = threadIdx.x;
  const int wave = tid >> 5;
  const int lane = tid & 31;
  const int n16  = lane & 15;
  const int hi   = lane >> 4;
  const int b    = blockIdx.x;

  // Stage V transposed (K-contiguous for the B operand).
  for (int idx = tid; idx < NN * ND; idx += 256) {
    const int j = idx >> 6, d = idx & 63;
    v_s[d * NN + j] = f2bf(v[((size_t)b * NN + j) * ND + d]);
  }

  // Softmax: wave handles 16 rows; 4 j's per lane; cross-lane reduce (wave32).
  for (int r = 0; r < 16; ++r) {
    const int i = wave * 16 + r;
    float sv[4];
#pragma unroll
    for (int c = 0; c < 4; ++c)
      sv[c] = scores[((size_t)(i * NN) + lane + 32 * c) * NB + b];
    float m = fmaxf(fmaxf(sv[0], sv[1]), fmaxf(sv[2], sv[3]));
#pragma unroll
    for (int o = 16; o >= 1; o >>= 1) m = fmaxf(m, __shfl_xor(m, o, 32));
    float ex[4], s = 0.0f;
#pragma unroll
    for (int c = 0; c < 4; ++c) { ex[c] = __expf(sv[c] - m); s += ex[c]; }
#pragma unroll
    for (int o = 16; o >= 1; o >>= 1) s += __shfl_xor(s, o, 32);
    const float inv = 1.0f / s;
#pragma unroll
    for (int c = 0; c < 4; ++c)
      a_s[i * NN + lane + 32 * c] = f2bf(ex[c] * inv);
  }
  __syncthreads();

  // out_b(128i x 64d) = A(128i x 128j) @ V(128j x 64d); wave -> one i-tile.
  const int ibase = wave * 16;
  v8f acc[4];
#pragma unroll
  for (int dt = 0; dt < 4; ++dt) acc[dt] = vzero();
#pragma unroll
  for (int js = 0; js < 4; ++js) {
    const int jb = js * 32;
    const v16bf afr = load_a(&a_s[(ibase + n16) * NN + jb], hi);
#pragma unroll
    for (int dt = 0; dt < 4; ++dt) {
      const v16bf bfr = load_b(&v_s[(dt * 16 + n16) * NN + jb], hi);
      acc[dt] = __builtin_amdgcn_wmma_f32_16x16x32_bf16(
          false, afr, false, bfr, (short)0, acc[dt], false, false);
    }
  }
  // C layout: M = r + hi*8, N = n16 -> 64B-coalesced f32 stores.
#pragma unroll
  for (int dt = 0; dt < 4; ++dt)
#pragma unroll
    for (int r = 0; r < 8; ++r) {
      const int i = ibase + r + hi * 8;
      out[((size_t)b * NN + i) * ND + dt * 16 + n16] = acc[dt][r];
    }
}

extern "C" void kernel_launch(void* const* d_in, const int* in_sizes, int n_in,
                              void* d_out, int out_size, void* d_ws,
                              size_t ws_size, hipStream_t stream) {
  const float* q   = (const float*)d_in[0];
  const float* k   = (const float*)d_in[1];
  const float* v   = (const float*)d_in[2];
  const float* mat = (const float*)d_in[3];
  float* out = (float*)d_out;
  float* l0  = out + (size_t)NB * NN * ND;  // scalar appended after (B,N,D)
  float* scores = (float*)d_ws;             // B*N*N f32 = 16 MB scratch

  hipLaunchKernelGGL(init_l0_kernel, dim3(1), dim3(32), 0, stream, l0);

  // Dynamic LDS: z_raw 32K + q 128K + k 128K + z 8K + u 16K + red 32 B.
  const size_t smem1 =
      32768 +
      (size_t)(4 * NB * ND + 4 * NB * ND + ND * ND + 8 * 16 * ND) * 2 +
      (size_t)8 * 4;  // = 319520 B (<= 320 KB/WGP)
  hipLaunchKernelGGL(score_gate_kernel, dim3(32 * 32), dim3(256), smem1,
                     stream, q, k, mat, scores, l0);

  hipLaunchKernelGGL(softmax_av_kernel, dim3(NB), dim3(256), 0, stream,
                     scores, v, out);
}
